// QMixer_82678120448731
// MI455X (gfx1250) — matrix operand order repaired and checked by reading, not compiled, not run on hardware
//
#include <hip/hip_runtime.h>
#include <math.h>

// ---------------------------------------------------------------------------
// QMIX mixer for MI455X (gfx1250, wave32, WMMA f16 -> f32 accumulate)
// ---------------------------------------------------------------------------

typedef _Float16 half_t;
typedef __attribute__((ext_vector_type(16))) _Float16 v16h;
typedef __attribute__((ext_vector_type(8)))  _Float16 v8h;
typedef __attribute__((ext_vector_type(8)))  float    v8f;
typedef __attribute__((ext_vector_type(4)))  float    v4f;

#define SDIM    512
#define WPITCH  136                     // LDS row pitch (halfs) = 128 + 8 pad
#define HPITCH  136
#define WBUF_HALFS (128 * WPITCH)       // weight slab buffer: 17408 halfs
#define HBUF_HALFS (8 * 16 * HPITCH)    // 8 waves x 16 rows x 136: 17408 halfs

__device__ __forceinline__ v8f wmma_f16(v16h a, v16h b, v8f c) {
  // v_wmma_f32_16x16x32_f16  (neg_a, A, neg_b, B, c_mod, C, reuse_a, reuse_b)
  return __builtin_amdgcn_wmma_f32_16x16x32_f16(false, a, false, b, (short)0, c,
                                                false, false);
}

__device__ __forceinline__ v16h pack16(v8h a, v8h b) {
  v16h r;
#pragma unroll
  for (int i = 0; i < 8; ++i) { r[i] = a[i]; r[8 + i] = b[i]; }
  return r;
}

// A fragment (16x32 f16) for row `row` of s (f32, ld=512), K-chunk base kb.
// Per ISA layout: lanes 0-15 hold K = kb+0..7 and kb+16..23, lanes 16-31
// hold K = kb+8..15 and kb+24..31  (hi = lane>=16 selects the +8 runs).
__device__ __forceinline__ v16h load_sA(const float* __restrict__ s, int row,
                                        int kb, int hi) {
  const float* p = s + (size_t)row * SDIM + kb + hi * 8;
  v4f f0 = *(const v4f*)(p);
  v4f f1 = *(const v4f*)(p + 4);
  v4f f2 = *(const v4f*)(p + 16);
  v4f f3 = *(const v4f*)(p + 20);
  v16h a;
#pragma unroll
  for (int i = 0; i < 4; ++i) {
    a[i]      = (half_t)f0[i];
    a[4 + i]  = (half_t)f1[i];
    a[8 + i]  = (half_t)f2[i];
    a[12 + i] = (half_t)f3[i];
  }
  return a;
}

// B fragment (32x16 f16) from LDS weight slab stored [n][k] (pitch WPITCH).
// Lane holds column n and K = chunk*32 + hi*8 + 0..7, +16..23 (koff = chunk*32 + hi*16).
__device__ __forceinline__ v16h load_B(const half_t* __restrict__ wbuf, int n,
                                       int koff) {
  const half_t* p = wbuf + n * WPITCH + (koff & ~31) + ((koff & 16) >> 1);
  return pack16(*(const v8h*)p, *(const v8h*)(p + 16));
}

// A fragment from per-wave h buffer (f16, [m][k], pitch HPITCH), K-chunk kc.
__device__ __forceinline__ v16h load_hA(const half_t* __restrict__ hbuf, int lo,
                                        int kc, int hi) {
  const half_t* p = hbuf + lo * HPITCH + kc * 32 + hi * 8;
  return pack16(*(const v8h*)p, *(const v8h*)(p + 16));
}

// GEMM: acc[NT tiles] = bias + s[16xSDIM] @ W[SDIM x NT*16], f16 WMMA.
// Weights staged to LDS (transposed to [n][k]) in K-slabs of 128, cooperatively
// by the whole workgroup (hence the barriers must be hit by all threads).
template <int NT, int LOG2N>
__device__ __forceinline__ void gemm_s(v8f (&acc)[NT],
                                       const float* __restrict__ s, int arow,
                                       const float* __restrict__ W,
                                       const float* __restrict__ bias,
                                       half_t* __restrict__ wbuf, int tid,
                                       int lo, int hi) {
  const int NC = NT * 16;
#pragma unroll
  for (int t = 0; t < NT; ++t) {
    float bv = bias[t * 16 + lo];
#pragma unroll
    for (int v = 0; v < 8; ++v) acc[t][v] = bv;
  }
  for (int slab = 0; slab < 4; ++slab) {
    __syncthreads();
    for (int i = tid; i < 128 * NC; i += 256) {
      int k = i >> LOG2N;
      int n = i & (NC - 1);
      wbuf[n * WPITCH + k] = (half_t)W[(size_t)(slab * 128 + k) * NC + n];
    }
    __syncthreads();
#pragma unroll
    for (int c = 0; c < 4; ++c) {
      v16h a = load_sA(s, arow, slab * 128 + c * 32, hi);
#pragma unroll
      for (int t = 0; t < NT; ++t) {
        v16h b = load_B(wbuf, t * 16 + lo, c * 32 + hi * 16);
        acc[t] = wmma_f16(a, b, acc[t]);
      }
    }
  }
}

__global__ void __launch_bounds__(256) qmix_kernel(
    const float* __restrict__ qs, const float* __restrict__ s,
    const float* __restrict__ W1a, const float* __restrict__ b1a,
    const float* __restrict__ W1b, const float* __restrict__ b1b,
    const float* __restrict__ Wb1, const float* __restrict__ bb1,
    const float* __restrict__ W2a, const float* __restrict__ b2a,
    const float* __restrict__ W2b, const float* __restrict__ b2b,
    const float* __restrict__ Wb2a, const float* __restrict__ bb2a,
    const float* __restrict__ Wb2b, const float* __restrict__ bb2b,
    float* __restrict__ out) {
  extern __shared__ half_t smem[];
  half_t* wbuf  = smem;                              // weight slab
  half_t* h1buf = smem + WBUF_HALFS;                 // per-wave h1 (f16)
  half_t* h2buf = smem + WBUF_HALFS + HBUF_HALFS;    // per-wave h2 (f16)

  const int tid  = threadIdx.x;
  const int lane = tid & 31;
  const int w    = tid >> 5;
  const int lo   = lane & 15;          // column / row-in-tile index
  const int hi   = lane >> 4;          // half-wave select
  const int rowBase = blockIdx.x * 128 + w * 16;
  const int arow = rowBase + lo;
  half_t* myh1 = h1buf + w * 16 * HPITCH;
  half_t* myh2 = h2buf + w * 16 * HPITCH;

  v16h h1A[4], h2A[4];

  // ---- h1 = relu(s @ W1a + b1a), kept as f16 A-fragments -----------------
  {
    v8f acc[8];
    gemm_s<8, 7>(acc, s, arow, W1a, b1a, wbuf, tid, lo, hi);
#pragma unroll
    for (int t = 0; t < 8; ++t)
#pragma unroll
      for (int v = 0; v < 8; ++v) {
        float x = acc[t][v];
        myh1[(v + 8 * hi) * HPITCH + t * 16 + lo] =
            (half_t)(x > 0.f ? x : 0.f);
      }
#pragma unroll
    for (int kc = 0; kc < 4; ++kc) h1A[kc] = load_hA(myh1, lo, kc, hi);
  }

  // ---- h2 = relu(s @ W2a + b2a) ------------------------------------------
  {
    v8f acc[8];
    gemm_s<8, 7>(acc, s, arow, W2a, b2a, wbuf, tid, lo, hi);
#pragma unroll
    for (int t = 0; t < 8; ++t)
#pragma unroll
      for (int v = 0; v < 8; ++v) {
        float x = acc[t][v];
        myh2[(v + 8 * hi) * HPITCH + t * 16 + lo] =
            (half_t)(x > 0.f ? x : 0.f);
      }
#pragma unroll
    for (int kc = 0; kc < 4; ++kc) h2A[kc] = load_hA(myh2, lo, kc, hi);
  }

  // ---- hid starts as b1 = s @ Wb1 + bb1 ----------------------------------
  v8f hid[4];
  gemm_s<4, 6>(hid, s, arow, Wb1, bb1, wbuf, tid, lo, hi);

  // ---- b2 = relu(s @ Wb2a + bb2a) @ Wb2b + bb2b  (scalar per row) --------
  float b2v[8];
  {
    v8f hb2[4];
    gemm_s<4, 6>(hb2, s, arow, Wb2a, bb2a, wbuf, tid, lo, hi);
    float p[8];
#pragma unroll
    for (int v = 0; v < 8; ++v) p[v] = 0.f;
#pragma unroll
    for (int t = 0; t < 4; ++t) {
      float wv = Wb2b[t * 16 + lo];
#pragma unroll
      for (int v = 0; v < 8; ++v) {
        float x = hb2[t][v];
        p[v] += (x > 0.f ? x : 0.f) * wv;
      }
    }
#pragma unroll
    for (int off = 1; off < 16; off <<= 1)
#pragma unroll
      for (int v = 0; v < 8; ++v) p[v] += __shfl_xor(p[v], off, 32);
    float bb = bb2b[0];
#pragma unroll
    for (int v = 0; v < 8; ++v) b2v[v] = p[v] + bb;
  }

  // ---- stage 2: w1 = |h1 @ W1b + b1b|, hid += q[a] * w1 tiles ------------
  // W1b is [128, 512]; process 4 N-slices of 128 cols, full K staged per slice.
  for (int slice = 0; slice < 4; ++slice) {
    __syncthreads();
    for (int i = tid; i < 128 * 128; i += 256) {
      int k = i >> 7;
      int n = i & 127;
      wbuf[n * WPITCH + k] = (half_t)W1b[(size_t)k * 512 + slice * 128 + n];
    }
    __syncthreads();
    // agents covered by this slice: a0 = slice*2 (tiles 0..3), a1 (tiles 4..7)
    float qf0[8], qf1[8];
    const int a0 = slice * 2;
#pragma unroll
    for (int v = 0; v < 8; ++v) {
      int r = rowBase + v + 8 * hi;
      qf0[v] = qs[r * 8 + a0];
      qf1[v] = qs[r * 8 + a0 + 1];
    }
#pragma unroll
    for (int t = 0; t < 8; ++t) {
      v8f acc;
      float bv = b1b[(slice * 8 + t) * 16 + lo];
#pragma unroll
      for (int v = 0; v < 8; ++v) acc[v] = bv;
#pragma unroll
      for (int c = 0; c < 4; ++c)
        acc = wmma_f16(h1A[c], load_B(wbuf, t * 16 + lo, c * 32 + hi * 16), acc);
      const int jt = t & 3;  // hidden column tile (64 = 4 tiles per agent)
#pragma unroll
      for (int v = 0; v < 8; ++v) {
        float qv = (t < 4) ? qf0[v] : qf1[v];
        hid[jt][v] += fabsf(acc[v]) * qv;
      }
    }
  }

  // ---- hidden = elu(hid) --------------------------------------------------
#pragma unroll
  for (int t = 0; t < 4; ++t)
#pragma unroll
    for (int v = 0; v < 8; ++v) {
      float x = hid[t][v];
      hid[t][v] = (x > 0.f) ? x : (__expf(x) - 1.f);
    }

  // ---- w2 = |h2 @ W2b + b2b| ---------------------------------------------
  v8f w2C[4];
  {
    __syncthreads();
    for (int i = tid; i < 128 * 64; i += 256) {
      int k = i >> 6;
      int n = i & 63;
      wbuf[n * WPITCH + k] = (half_t)W2b[k * 64 + n];
    }
    __syncthreads();
#pragma unroll
    for (int t = 0; t < 4; ++t) {
      v8f acc;
      float bv = b2b[t * 16 + lo];
#pragma unroll
      for (int v = 0; v < 8; ++v) acc[v] = bv;
#pragma unroll
      for (int c = 0; c < 4; ++c)
        acc = wmma_f16(h2A[c], load_B(wbuf, t * 16 + lo, c * 32 + hi * 16), acc);
#pragma unroll
      for (int v = 0; v < 8; ++v) w2C[t][v] = fabsf(acc[v]);
    }
  }

  // ---- q_total = hidden . w2 + b2 ----------------------------------------
  {
    float p[8];
#pragma unroll
    for (int v = 0; v < 8; ++v) p[v] = 0.f;
#pragma unroll
    for (int t = 0; t < 4; ++t)
#pragma unroll
      for (int v = 0; v < 8; ++v) p[v] += hid[t][v] * w2C[t][v];
#pragma unroll
    for (int off = 1; off < 16; off <<= 1)
#pragma unroll
      for (int v = 0; v < 8; ++v) p[v] += __shfl_xor(p[v], off, 32);
    // C-layout: vgpr v, half-wave hi holds row v + 8*hi; lane lo==v stores it.
#pragma unroll
    for (int v = 0; v < 8; ++v)
      if (lo == v) out[rowBase + 8 * hi + v] = p[v] + b2v[v];
  }
}

extern "C" void kernel_launch(void* const* d_in, const int* in_sizes, int n_in,
                              void* d_out, int out_size, void* d_ws,
                              size_t ws_size, hipStream_t stream) {
  (void)in_sizes; (void)n_in; (void)out_size; (void)d_ws; (void)ws_size;
  const float* qs   = (const float*)d_in[0];
  const float* s    = (const float*)d_in[1];
  const float* W1a  = (const float*)d_in[2];
  const float* b1a  = (const float*)d_in[3];
  const float* W1b  = (const float*)d_in[4];
  const float* b1b  = (const float*)d_in[5];
  const float* Wb1  = (const float*)d_in[6];
  const float* bb1  = (const float*)d_in[7];
  const float* W2a  = (const float*)d_in[8];
  const float* b2a  = (const float*)d_in[9];
  const float* W2b  = (const float*)d_in[10];
  const float* b2b  = (const float*)d_in[11];
  const float* Wb2a = (const float*)d_in[12];
  const float* bb2a = (const float*)d_in[13];
  const float* Wb2b = (const float*)d_in[14];
  const float* bb2b = (const float*)d_in[15];
  float* out = (float*)d_out;

  const size_t shmem = (size_t)(WBUF_HALFS + 2 * HBUF_HALFS) * sizeof(half_t);
  dim3 grid(131072 / 128);   // 128 batch rows per workgroup (8 waves x 16)
  dim3 block(256);
  qmix_kernel<<<grid, block, shmem, stream>>>(qs, s, W1a, b1a, W1b, b1b, Wb1,
                                              bb1, W2a, b2a, W2b, b2b, Wb2a,
                                              bb2a, Wb2b, bb2b, out);
}
